// MultiHeadedSelfAttention_27109833573012
// MI455X (gfx1250) — matrix operand order repaired and checked by reading, not compiled, not run on hardware
//
#include <hip/hip_runtime.h>

// ---------------------------------------------------------------------------
// MI455X (gfx1250) multi-head self-attention: B=2, S=2048, D=1024, H=16, DK=64
// GEMMs via v_wmma_f32_16x16x32_bf16; tile staging via Tensor Data Mover
// (tensor_load_to_lds + s_wait_tensorcnt), double-buffered LDS.
// ---------------------------------------------------------------------------

typedef __bf16 bf16;
typedef __attribute__((ext_vector_type(4)))  bf16  v4bf;
typedef __attribute__((ext_vector_type(8)))  bf16  v8bf;
typedef __attribute__((ext_vector_type(16))) bf16  v16bf;
typedef __attribute__((ext_vector_type(8)))  float v8f;
typedef __attribute__((ext_vector_type(4)))  unsigned int u32x4;
typedef __attribute__((ext_vector_type(8)))  int i32x8;
typedef __attribute__((ext_vector_type(4)))  int i32x4;

#define B_  2
#define S_  2048
#define D_  1024
#define H_  16
#define DK_ 64
#define M_  (B_ * S_)   // 4096 total rows
#define LD_ 40          // padded LDS row stride (elements) = 32 + 16B TDM pad

static __device__ __forceinline__ bf16 f2bf(float f) {
  unsigned u = __builtin_bit_cast(unsigned, f);
  u += 0x7FFFu + ((u >> 16) & 1u);                 // round-to-nearest-even
  unsigned short h = (unsigned short)(u >> 16);
  return __builtin_bit_cast(bf16, h);
}
static __device__ __forceinline__ v8f v8f_zero() {
  v8f z;
#pragma unroll
  for (int i = 0; i < 8; ++i) z[i] = 0.0f;
  return z;
}
static __device__ __forceinline__ v16bf mk16(v8bf lo, v8bf hi) {
  v16bf r;
#pragma unroll
  for (int i = 0; i < 8; ++i) { r[i] = lo[i]; r[i + 8] = hi[i]; }
  return r;
}
#define WMMA_BF16(a, b, c) \
  __builtin_amdgcn_wmma_f32_16x16x32_bf16(false, (a), false, (b), (short)0, (c), false, false)

// Low 32 bits of a generic pointer to __shared__ == LDS byte offset (ISA 10.2).
static __device__ __forceinline__ unsigned lds_off(const void* p) {
  return (unsigned)(size_t)p;
}

// ---------------------------------------------------------------------------
// TDM: DMA a 2-D bf16 tile [tile_rows x 32] (row stride = row_stride_elems in
// global) into LDS, with 16B padding after every 64B row -> LDS stride 40 elems.
// Descriptor per CDNA5 ISA ch.8 (D# groups 0/1; groups 2/3 zero for 2-D).
// Must be called under uniform control flow (wave-level op, EXEC ignored).
// ---------------------------------------------------------------------------
static __device__ __forceinline__ void tdm_load_tile_bf16(
    const bf16* gptr, unsigned ldsaddr, int tile_rows, int tensor_rows,
    int row_stride_elems) {
  const unsigned long long ga = (unsigned long long)(size_t)gptr;
  u32x4 g0;
  g0[0] = 1u;                                            // count=1, no gather
  g0[1] = ldsaddr;                                       // lds_addr
  g0[2] = (unsigned)(ga & 0xFFFFFFFFu);                  // global_addr[31:0]
  g0[3] = (unsigned)((ga >> 32) & 0x01FFFFFFu) | (2u << 30);  // addr[56:32]|type=2
  i32x8 g1;
  g1[0] = (1 << 16)        // data_size = 2 bytes
        | (1 << 20)        // pad_enable
        | (3 << 22)        // pad_interval: 16 DWORDs (64B row)
        | (3 << 25);       // pad_amount:   4 DWORDs (16B)
  g1[1] = (row_stride_elems & 0xFFFF) << 16;             // tensor_dim0[15:0]
  g1[2] = ((row_stride_elems >> 16) & 0xFFFF)            // tensor_dim0[31:16]
        | ((tensor_rows & 0xFFFF) << 16);                // tensor_dim1[15:0]
  g1[3] = ((tensor_rows >> 16) & 0xFFFF)                 // tensor_dim1[31:16]
        | (32 << 16);                                    // tile_dim0 = 32 elems
  g1[4] = tile_rows & 0xFFFF;                            // tile_dim1 (dim2=0)
  g1[5] = row_stride_elems;                              // dim0_stride[31:0]
  g1[6] = 0;                                             // dim0_stride[47:32], dim1_stride[15:0]
  g1[7] = 0;                                             // dim1_stride[47:16]
  i32x4 z4; z4[0] = 0; z4[1] = 0; z4[2] = 0; z4[3] = 0;
  i32x8 z8;
#pragma unroll
  for (int i = 0; i < 8; ++i) z8[i] = 0;
  // 6-arg form (this toolchain): (g0, g1, g2, g3, extra, cpol)
  __builtin_amdgcn_tensor_load_to_lds(g0, g1, z4, z4, z8, 0);
}

// ---------------------------------------------------------------------------
// Kernel 0: activation fp32 -> bf16 (so TDM can stage X tiles directly)
// ---------------------------------------------------------------------------
__global__ __launch_bounds__(256) void xcvt_kernel(const float* __restrict__ X,
                                                   bf16* __restrict__ Xb) {
  const int i = (blockIdx.x * 256 + threadIdx.x) * 4;
  float4 f = *reinterpret_cast<const float4*>(&X[i]);
  v4bf h; h[0] = f2bf(f.x); h[1] = f2bf(f.y); h[2] = f2bf(f.z); h[3] = f2bf(f.w);
  *reinterpret_cast<v4bf*>(&Xb[i]) = h;
}

// ---------------------------------------------------------------------------
// Kernel 1: weight fp32 [k][n] -> bf16 transposed [n][k] (B-fragment friendly)
// ---------------------------------------------------------------------------
__global__ __launch_bounds__(256) void wtrans_kernel(const float* __restrict__ W,
                                                     bf16* __restrict__ WT) {
  const int i = blockIdx.x * 256 + threadIdx.x;   // coalesced read along n
  const int k = i >> 10;
  const int n = i & (D_ - 1);
  WT[(size_t)n * D_ + k] = f2bf(W[i]);
}

// ---------------------------------------------------------------------------
// Kernel 2: fused Q/K/V projection, TDM double-buffered.
// Block = 8 waves; tile = 128(M) x 64(N); K-step = 32. Wave 0 drives the TDM
// pipeline (4 tile DMAs/step), all waves compute 12 WMMAs/step.
// Q,K stored [B,H,S,DK]; V stored transposed [B,H,DK,S].
// ---------------------------------------------------------------------------
__global__ __launch_bounds__(256) void qkv_proj_kernel(
    const bf16* __restrict__ Xb,
    const bf16* __restrict__ WqT, const bf16* __restrict__ WkT,
    const bf16* __restrict__ WvT,
    const float* __restrict__ bq, const float* __restrict__ bk,
    const float* __restrict__ bv,
    bf16* __restrict__ qbuf, bf16* __restrict__ kbuf, bf16* __restrict__ vbufT) {
  const int m0   = blockIdx.x * 128;
  const int n0   = blockIdx.y * 64;
  const int tid  = threadIdx.x;
  const int wave = tid >> 5;
  const int lane = tid & 31;
  const int lrow = lane & 15;
  const int lhi  = lane >> 4;

  __shared__ bf16 Xs[2][128 * LD_];        // double-buffered A tiles
  __shared__ bf16 Wls[2][3][64 * LD_];     // double-buffered B tiles x3

  v8f accQ[4], accK[4], accV[4];
#pragma unroll
  for (int j = 0; j < 4; ++j) { accQ[j] = v8f_zero(); accK[j] = v8f_zero(); accV[j] = v8f_zero(); }

  const int NSTEP = D_ / 32;

  if (wave == 0) {   // uniform branch: wave-level TDM issue
    tdm_load_tile_bf16(Xb  + (size_t)m0 * D_, lds_off(&Xs[0][0]),     128, M_, D_);
    tdm_load_tile_bf16(WqT + (size_t)n0 * D_, lds_off(&Wls[0][0][0]),  64, D_, D_);
    tdm_load_tile_bf16(WkT + (size_t)n0 * D_, lds_off(&Wls[0][1][0]),  64, D_, D_);
    tdm_load_tile_bf16(WvT + (size_t)n0 * D_, lds_off(&Wls[0][2][0]),  64, D_, D_);
  }

  for (int it = 0; it < NSTEP; ++it) {
    const int buf = it & 1;
    if (wave == 0) {
      if (it + 1 < NSTEP) {
        const int kc = (it + 1) * 32;
        tdm_load_tile_bf16(Xb  + (size_t)m0 * D_ + kc, lds_off(&Xs[buf ^ 1][0]),     128, M_, D_);
        tdm_load_tile_bf16(WqT + (size_t)n0 * D_ + kc, lds_off(&Wls[buf ^ 1][0][0]),  64, D_, D_);
        tdm_load_tile_bf16(WkT + (size_t)n0 * D_ + kc, lds_off(&Wls[buf ^ 1][1][0]),  64, D_, D_);
        tdm_load_tile_bf16(WvT + (size_t)n0 * D_ + kc, lds_off(&Wls[buf ^ 1][2][0]),  64, D_, D_);
        __builtin_amdgcn_s_wait_tensorcnt(4);   // current step's 4 DMAs done
      } else {
        __builtin_amdgcn_s_wait_tensorcnt(0);
      }
    }
    __syncthreads();

    // A fragment: lane holds row (wave*16+lrow), K = lhi*16+0..15
    const bf16* ap = &Xs[buf][(wave * 16 + lrow) * LD_ + lhi * 16];
    v16bf aX = mk16(*(const v8bf*)ap, *(const v8bf*)(ap + 8));

#pragma unroll
    for (int j = 0; j < 4; ++j) {
      const int boff = (j * 16 + lrow) * LD_ + lhi * 16;  // B: col=lrow, K=lhi*16+i
      v16bf bQ = mk16(*(const v8bf*)&Wls[buf][0][boff], *(const v8bf*)&Wls[buf][0][boff + 8]);
      v16bf bK = mk16(*(const v8bf*)&Wls[buf][1][boff], *(const v8bf*)&Wls[buf][1][boff + 8]);
      v16bf bV = mk16(*(const v8bf*)&Wls[buf][2][boff], *(const v8bf*)&Wls[buf][2][boff + 8]);
      accQ[j] = WMMA_BF16(aX, bQ, accQ[j]);
      accK[j] = WMMA_BF16(aX, bK, accK[j]);
      accV[j] = WMMA_BF16(aX, bV, accV[j]);
    }
    __syncthreads();
  }

  // Epilogue: bias + bf16 store to head-major layouts (C-layout: M=lhi*8+v, N=lrow)
#pragma unroll
  for (int j = 0; j < 4; ++j) {
    const int coln = n0 + j * 16 + lrow;
    const int hh = coln >> 6;
    const int dd = coln & (DK_ - 1);
    const float bqv = bq[coln], bkv = bk[coln], bvv = bv[coln];
#pragma unroll
    for (int v = 0; v < 8; ++v) {
      const int grow = m0 + wave * 16 + lhi * 8 + v;
      const int bb = grow >> 11;           // / S_
      const int ss = grow & (S_ - 1);
      const size_t qk = (((size_t)bb * H_ + hh) * S_ + ss) * DK_ + dd;
      qbuf[qk] = f2bf(accQ[j][v] + bqv);
      kbuf[qk] = f2bf(accK[j][v] + bkv);
      vbufT[(((size_t)bb * H_ + hh) * DK_ + dd) * S_ + ss] = f2bf(accV[j][v] + bvv);
    }
  }
}

// ---------------------------------------------------------------------------
// Kernel 3: flash-style attention. Grid = (S/128, B*H); 8 waves, each owns 16
// query rows. Online softmax in registers; LDS only for the C->A layout
// transpose of P. global_prefetch for next K/V chunk.
// ---------------------------------------------------------------------------
__global__ __launch_bounds__(256) void attn_kernel(
    const bf16* __restrict__ qb, const bf16* __restrict__ kb,
    const bf16* __restrict__ vT, bf16* __restrict__ cb) {
  const int q0   = blockIdx.x * 128;
  const int bh   = blockIdx.y;
  const int bIdx = bh >> 4;
  const int h    = bh & (H_ - 1);
  const int tid  = threadIdx.x;
  const int wave = tid >> 5;
  const int lane = tid & 31;
  const int lrow = lane & 15;
  const int lhi  = lane >> 4;

  __shared__ bf16 Plds[8][16 * LD_];    // per-wave 16x32 P tile

  const bf16* Qh = qb + (size_t)bh * S_ * DK_;
  const bf16* Kh = kb + (size_t)bh * S_ * DK_;
  const bf16* Vh = vT + (size_t)bh * DK_ * S_;   // [DK][S]

  const int qrow = q0 + wave * 16 + lrow;
  v16bf aQ[2];
#pragma unroll
  for (int c = 0; c < 2; ++c) {
    const bf16* p = &Qh[(size_t)qrow * DK_ + c * 32 + lhi * 16];
    aQ[c] = mk16(*(const v8bf*)p, *(const v8bf*)(p + 8));
  }

  v8f o[4];
#pragma unroll
  for (int n = 0; n < 4; ++n) o[n] = v8f_zero();
  float mrun[8], lsum[8];
#pragma unroll
  for (int v = 0; v < 8; ++v) { mrun[v] = -1e30f; lsum[v] = 0.0f; }

  for (int kc = 0; kc < S_; kc += 32) {
    // prefetch next chunk of K (32 rows x 128B) and V^T (64 rows' next 64B)
    if (kc + 32 < S_) {
      __builtin_prefetch(&Kh[(size_t)(kc + 32 + lane) * DK_], 0, 1);
      __builtin_prefetch(&Vh[(size_t)lane * S_ + kc + 32], 0, 1);
      __builtin_prefetch(&Vh[(size_t)(32 + lane) * S_ + kc + 32], 0, 1);
    }

    // ---- scores S = Q K^T * 1/sqrt(DK): 2 N-tiles x 2 K-steps ----
    v8f sa[2];
#pragma unroll
    for (int j = 0; j < 2; ++j) {
      const int scol = kc + j * 16 + lrow;           // B col = key position
      const bf16* kp = &Kh[(size_t)scol * DK_ + lhi * 16];
      v16bf b0 = mk16(*(const v8bf*)kp,        *(const v8bf*)(kp + 8));   // d 0..31
      v16bf b1 = mk16(*(const v8bf*)(kp + 32), *(const v8bf*)(kp + 40));  // d 32..63
      v8f acc = v8f_zero();
      acc = WMMA_BF16(aQ[0], b0, acc);
      acc = WMMA_BF16(aQ[1], b1, acc);
#pragma unroll
      for (int v = 0; v < 8; ++v) acc[v] *= 0.125f;  // 1/sqrt(64)
      sa[j] = acc;
    }

    // ---- chunk row-max (reduce across the 16-lane N group) ----
    float pm[8];
#pragma unroll
    for (int v = 0; v < 8; ++v) pm[v] = fmaxf(sa[0][v], sa[1][v]);
#pragma unroll
    for (int m = 1; m <= 8; m <<= 1)
#pragma unroll
      for (int v = 0; v < 8; ++v) pm[v] = fmaxf(pm[v], __shfl_xor(pm[v], m, 32));

    float alpha[8];
#pragma unroll
    for (int v = 0; v < 8; ++v) {
      const float mn = fmaxf(mrun[v], pm[v]);
      alpha[v] = __expf(mrun[v] - mn);
      mrun[v] = mn;
    }

    // ---- P = exp(S - m); row sums; spill P to LDS in bf16 ----
    float psum[8];
#pragma unroll
    for (int v = 0; v < 8; ++v) psum[v] = 0.0f;
#pragma unroll
    for (int j = 0; j < 2; ++j)
#pragma unroll
      for (int v = 0; v < 8; ++v) {
        const float p = __expf(sa[j][v] - mrun[v]);
        psum[v] += p;
        Plds[wave][(lhi * 8 + v) * LD_ + j * 16 + lrow] = f2bf(p);
      }
#pragma unroll
    for (int m = 1; m <= 8; m <<= 1)
#pragma unroll
      for (int v = 0; v < 8; ++v) psum[v] += __shfl_xor(psum[v], m, 32);
#pragma unroll
    for (int v = 0; v < 8; ++v) lsum[v] = lsum[v] * alpha[v] + psum[v];
#pragma unroll
    for (int n = 0; n < 4; ++n)
#pragma unroll
      for (int v = 0; v < 8; ++v) o[n][v] *= alpha[v];

    // ---- O += P V (same-wave LDS RAW is in-order; V^T rows contiguous) ----
    const bf16* pp = &Plds[wave][lrow * LD_ + lhi * 16];
    v16bf aP = mk16(*(const v8bf*)pp, *(const v8bf*)(pp + 8));
#pragma unroll
    for (int n = 0; n < 4; ++n) {
      const bf16* vp = &Vh[(size_t)(n * 16 + lrow) * S_ + kc + lhi * 16];
      v16bf bV = mk16(*(const v8bf*)vp, *(const v8bf*)(vp + 8));
      o[n] = WMMA_BF16(aP, bV, o[n]);
    }
  }

  // ---- normalize and store concat buffer [B,S,D] in bf16 ----
#pragma unroll
  for (int n = 0; n < 4; ++n)
#pragma unroll
    for (int v = 0; v < 8; ++v) {
      const int srow = q0 + wave * 16 + lhi * 8 + v;
      const int col  = h * DK_ + n * 16 + lrow;
      cb[((size_t)bIdx * S_ + srow) * D_ + col] = f2bf(o[n][v] / lsum[v]);
    }
}

// ---------------------------------------------------------------------------
// Kernel 4: output projection, TDM double-buffered, f32 epilogue into d_out.
// ---------------------------------------------------------------------------
__global__ __launch_bounds__(256) void oproj_kernel(
    const bf16* __restrict__ Xb, const bf16* __restrict__ WoT,
    const float* __restrict__ bo, float* __restrict__ out) {
  const int m0   = blockIdx.x * 128;
  const int n0   = blockIdx.y * 64;
  const int tid  = threadIdx.x;
  const int wave = tid >> 5;
  const int lane = tid & 31;
  const int lrow = lane & 15;
  const int lhi  = lane >> 4;

  __shared__ bf16 Xs[2][128 * LD_];
  __shared__ bf16 Ws[2][64 * LD_];

  v8f acc[4];
#pragma unroll
  for (int j = 0; j < 4; ++j) acc[j] = v8f_zero();

  const int NSTEP = D_ / 32;
  if (wave == 0) {
    tdm_load_tile_bf16(Xb  + (size_t)m0 * D_, lds_off(&Xs[0][0]), 128, M_, D_);
    tdm_load_tile_bf16(WoT + (size_t)n0 * D_, lds_off(&Ws[0][0]),  64, D_, D_);
  }

  for (int it = 0; it < NSTEP; ++it) {
    const int buf = it & 1;
    if (wave == 0) {
      if (it + 1 < NSTEP) {
        const int kc = (it + 1) * 32;
        tdm_load_tile_bf16(Xb  + (size_t)m0 * D_ + kc, lds_off(&Xs[buf ^ 1][0]), 128, M_, D_);
        tdm_load_tile_bf16(WoT + (size_t)n0 * D_ + kc, lds_off(&Ws[buf ^ 1][0]),  64, D_, D_);
        __builtin_amdgcn_s_wait_tensorcnt(2);
      } else {
        __builtin_amdgcn_s_wait_tensorcnt(0);
      }
    }
    __syncthreads();

    const bf16* ap = &Xs[buf][(wave * 16 + lrow) * LD_ + lhi * 16];
    v16bf aX = mk16(*(const v8bf*)ap, *(const v8bf*)(ap + 8));
#pragma unroll
    for (int j = 0; j < 4; ++j) {
      const int boff = (j * 16 + lrow) * LD_ + lhi * 16;
      v16bf bW = mk16(*(const v8bf*)&Ws[buf][boff], *(const v8bf*)&Ws[buf][boff + 8]);
      acc[j] = WMMA_BF16(aX, bW, acc[j]);
    }
    __syncthreads();
  }

#pragma unroll
  for (int j = 0; j < 4; ++j) {
    const int coln = n0 + j * 16 + lrow;
    const float bias = bo[coln];
#pragma unroll
    for (int v = 0; v < 8; ++v) {
      const int grow = m0 + wave * 16 + lhi * 8 + v;
      out[(size_t)grow * D_ + coln] = acc[j][v] + bias;
    }
  }
}

// ---------------------------------------------------------------------------
extern "C" void kernel_launch(void* const* d_in, const int* in_sizes, int n_in,
                              void* d_out, int out_size, void* d_ws, size_t ws_size,
                              hipStream_t stream) {
  (void)in_sizes; (void)n_in; (void)out_size; (void)ws_size;
  const float* X  = (const float*)d_in[0];
  const float* Wq = (const float*)d_in[1];
  const float* bq = (const float*)d_in[2];
  const float* Wk = (const float*)d_in[3];
  const float* bk = (const float*)d_in[4];
  const float* Wv = (const float*)d_in[5];
  const float* bv = (const float*)d_in[6];
  const float* Wo = (const float*)d_in[7];
  const float* bo = (const float*)d_in[8];
  float* out = (float*)d_out;

  char* p = (char*)d_ws;
  const size_t wbytes = (size_t)D_ * D_ * sizeof(bf16);   // 2 MB each
  const size_t abytes = (size_t)M_ * D_ * sizeof(bf16);   // 8 MB each
  bf16* WqT = (bf16*)p; p += wbytes;
  bf16* WkT = (bf16*)p; p += wbytes;
  bf16* WvT = (bf16*)p; p += wbytes;
  bf16* WoT = (bf16*)p; p += wbytes;
  bf16* Xb  = (bf16*)p; p += abytes;
  bf16* qb  = (bf16*)p; p += abytes;
  bf16* kb  = (bf16*)p; p += abytes;
  bf16* vT  = (bf16*)p; p += abytes;
  bf16* cb  = (bf16*)p; p += abytes;     // total ~48 MB

  const int tgrid = (D_ * D_) / 256;
  xcvt_kernel<<<(M_ * D_) / (256 * 4), 256, 0, stream>>>(X, Xb);
  wtrans_kernel<<<tgrid, 256, 0, stream>>>(Wq, WqT);
  wtrans_kernel<<<tgrid, 256, 0, stream>>>(Wk, WkT);
  wtrans_kernel<<<tgrid, 256, 0, stream>>>(Wv, WvT);
  wtrans_kernel<<<tgrid, 256, 0, stream>>>(Wo, WoT);

  qkv_proj_kernel<<<dim3(M_ / 128, D_ / 64), 256, 0, stream>>>(
      Xb, WqT, WkT, WvT, bq, bk, bv, qb, kb, vT);
  attn_kernel<<<dim3(S_ / 128, B_ * H_), 256, 0, stream>>>(qb, kb, vT, cb);
  oproj_kernel<<<dim3(M_ / 128, D_ / 64), 256, 0, stream>>>(cb, WoT, bo, out);
}
